// RiemannianGraphVAE_38826504356648
// MI455X (gfx1250) — matrix operand order repaired and checked by reading, not compiled, not run on hardware
//
#include <hip/hip_runtime.h>
#include <hip/hip_bf16.h>

typedef __attribute__((ext_vector_type(16))) _Float16 v16h;
typedef __attribute__((ext_vector_type(8)))  _Float16 v8h;
typedef __attribute__((ext_vector_type(8)))  float    v8f;

union V16 { v16h v; v8h h[2]; };

#define NN 512
#define BB 8
#define FF 16
#define HH 64
#define LL 32
#define PAIRS 130816      /* 512*511/2 */
#define PT16  8176        /* PAIRS/16  */
#define ADJ_SZ (BB*NN*NN)
#define MU_SZ  (BB*NN*LL)

static __device__ __forceinline__ v8f wmma_f16(const V16& a, const V16& b, v8f c) {
  return __builtin_amdgcn_wmma_f32_16x16x32_f16(false, a.v, false, b.v, (short)0, c, false, false);
}

// ---- pair index table: p -> (i, j) for upper triangle ----
__global__ void k_pairs(int* iu, int* ju) {
  int i = blockIdx.x, j = threadIdx.x;
  if (j > i) {
    int off = i * (NN - 1) - (i * (i - 1)) / 2;
    int p = off + (j - i - 1);
    iu[p] = i; ju[p] = j;
  }
}

// ---- adjacency f32 -> f16 (values are exact 0/1) ----
__global__ void k_adj_half(const float* __restrict__ adj, _Float16* __restrict__ adjh, int n) {
  int t = blockIdx.x * blockDim.x + threadIdx.x;
  if (t < n) adjh[t] = (_Float16)adj[t];
}

// ---- degree (with self loop) -> dinv = 1/sqrt(deg) ----
__global__ void k_deg(const float* __restrict__ adj, float* __restrict__ dinv) {
  int t = blockIdx.x * blockDim.x + threadIdx.x;
  if (t < BB * NN) {
    const float* row = adj + (size_t)t * NN;
    float s = 1.0f;
    for (int j = 0; j < NN; ++j) s += row[j];
    dinv[t] = rsqrtf(s);
  }
}

// ---- decoder weights, transposed + f16 (B-tiles become contiguous loads) ----
__global__ void k_wcvt(const float* __restrict__ We1, const float* __restrict__ We2,
                       _Float16* __restrict__ We1t, _Float16* __restrict__ We2t) {
  int t = blockIdx.x * blockDim.x + threadIdx.x;
  if (t < 128 * 64) {                    // We1t[n<128][k<64] = We1[k][n]
    int n = t / 64, k = t % 64;
    We1t[n * 64 + k] = (_Float16)We1[k * 128 + n];
  } else if (t < 2 * 128 * 64) {         // We2t[n<64][k<128] = We2[k][n]
    int q = t - 128 * 64;
    int n = q / 128, k = q % 128;
    We2t[n * 128 + k] = (_Float16)We2[k * 64 + n];
  }
}

// ---- t[b][h][j] = f16( dinv_j * (x_j . W[:,h]) ), stored transposed ----
__global__ void k_xw(const float* __restrict__ in, int K, int batchStride,
                     const float* __restrict__ W, const float* __restrict__ dinv,
                     _Float16* __restrict__ th) {
  int t = blockIdx.x * blockDim.x + threadIdx.x;
  if (t >= BB * HH * NN) return;
  int b = t / (HH * NN); int rem = t % (HH * NN);
  int h = rem / NN;      int j = rem % NN;
  const float* row = in + (size_t)b * batchStride + (size_t)j * K;
  float s = 0.f;
  for (int f = 0; f < K; ++f) s += row[f] * W[f * HH + h];
  th[((size_t)b * HH + h) * NN + j] = (_Float16)(dinv[b * NN + j] * s);
}

// ---- GCN aggregation via WMMA: out = relu(dinv_i*(adj@t + t_i) + bias) [+ res] ----
__global__ void k_gcn_agg(const _Float16* __restrict__ adjh, const _Float16* __restrict__ th,
                          const float* __restrict__ dinv, const float* __restrict__ bias,
                          const float* __restrict__ res, float* __restrict__ out) {
  int lane = threadIdx.x & 31, wv = threadIdx.x >> 5;
  int tile = blockIdx.x * 8 + wv;          // 1024 tiles total, grid = 128 blocks
  int b  = tile >> 7;
  int rm = tile & 127;
  int it = rm >> 2, ht = rm & 3;
  int lo = lane & 15, hi = lane >> 4;
  const _Float16* arow = adjh + ((size_t)b * NN + it * 16 + lo) * NN;
  const _Float16* brow = th   + ((size_t)b * HH + ht * 16 + lo) * NN;
  v8f c = {};
  for (int kk = 0; kk < NN / 32; ++kk) {
    int k0 = kk * 32;
    V16 a, bm;
    a.h[0]  = *(const v8h*)(arow + k0 + 8 * hi);
    a.h[1]  = *(const v8h*)(arow + k0 + 16 + 8 * hi);
    bm.h[0] = *(const v8h*)(brow + k0 + 8 * hi);
    bm.h[1] = *(const v8h*)(brow + k0 + 16 + 8 * hi);
    c = wmma_f16(a, bm, c);
  }
  int h = ht * 16 + lo;
  float bv = bias[h];
  #pragma unroll
  for (int r = 0; r < 8; ++r) {
    int i = it * 16 + 8 * hi + r;
    float ti = (float)th[((size_t)b * HH + h) * NN + i];  // self-loop term
    float v = dinv[b * NN + i] * (c[r] + ti) + bv;
    v = v > 0.f ? v : 0.f;
    size_t o = ((size_t)b * NN + i) * HH + h;
    if (res) v += res[o];
    out[o] = v;
  }
}

// ---- latent heads: mu / logvar / zlat (+ f16 copy for decoder) ----
__global__ void k_mu(const float* __restrict__ h3, const float* __restrict__ Wmu,
                     const float* __restrict__ bmu, const float* __restrict__ Wlv,
                     const float* __restrict__ blv, float* __restrict__ out,
                     _Float16* __restrict__ zh) {
  int t = blockIdx.x * blockDim.x + threadIdx.x;
  if (t >= MU_SZ) return;
  int bn = t / LL, l = t % LL;
  const float* hr = h3 + (size_t)bn * HH;
  float sm = bmu[l], sv = blv[l];
  for (int h = 0; h < HH; ++h) { float x = hr[h]; sm += x * Wmu[h * LL + l]; sv += x * Wlv[h * LL + l]; }
  out[ADJ_SZ + t]             = sm;
  out[ADJ_SZ + MU_SZ + t]     = sv;
  out[ADJ_SZ + 2 * MU_SZ + t] = sm;   // zlat == mu in eval mode
  zh[t] = (_Float16)sm;
}

__global__ void k_zero_diag(float* __restrict__ out) {
  int t = blockIdx.x * blockDim.x + threadIdx.x;
  if (t < BB * NN) { int b = t / NN, i = t % NN; out[(size_t)b * NN * NN + i * NN + i] = 0.f; }
}

// ---- edge decoder MLP: one wave per 16-pair tile, f16 WMMA, f32 accumulate ----
__global__ __launch_bounds__(128) void k_decode(
    const _Float16* __restrict__ zh, const _Float16* __restrict__ We1t,
    const _Float16* __restrict__ We2t, const float* __restrict__ be1,
    const float* __restrict__ be2, const float* __restrict__ We3,
    const float* __restrict__ be3, const int* __restrict__ iu,
    const int* __restrict__ ju, float* __restrict__ out) {
  __shared__ _Float16 sH1[4][16 * 128];   // per-wave private C->A transpose staging
  __shared__ float sBe1[128];
  __shared__ float sBe2[64];
  __shared__ float sWe3[64];
  int tid = threadIdx.x;
  if (tid < 128) sBe1[tid] = be1[tid];
  if (tid < 64) { sBe2[tid] = be2[tid]; sWe3[tid] = We3[tid]; }
  __syncthreads();
  float be3v = be3[0];
  int lane = tid & 31, wv = tid >> 5;
  int lo = lane & 15, hi = lane >> 4;
  int wid = blockIdx.x * 4 + wv;
  int nw  = gridDim.x * 4;
  _Float16* h1 = sH1[wv];
  for (int tile = wid; tile < BB * PT16; tile += nw) {
    int b = tile / PT16, pt = tile % PT16;
    int p = pt * 16 + lo;                 // A-row pair for this lane
    int i = iu[p], j = ju[p];
    const _Float16* zi = zh + ((size_t)b * NN + i) * LL;
    const _Float16* zj = zh + ((size_t)b * NN + j) * LL;
    // ---- layer 1: [16x64] @ [64x128] ----
    v8f c1[8];
    #pragma unroll
    for (int tn = 0; tn < 8; ++tn)
      #pragma unroll
      for (int r = 0; r < 8; ++r) c1[tn][r] = 0.f;
    #pragma unroll
    for (int ks = 0; ks < 2; ++ks) {      // ks=0: z_i half, ks=1: z_j half (K split at 32)
      const _Float16* zr = ks ? zj : zi;
      V16 a;
      a.h[0] = *(const v8h*)(zr + 8 * hi);
      a.h[1] = *(const v8h*)(zr + 16 + 8 * hi);
      #pragma unroll
      for (int tn = 0; tn < 8; ++tn) {
        const _Float16* wr = We1t + (tn * 16 + lo) * 64 + ks * 32;
        V16 bm;
        bm.h[0] = *(const v8h*)(wr + 8 * hi);
        bm.h[1] = *(const v8h*)(wr + 16 + 8 * hi);
        c1[tn] = wmma_f16(a, bm, c1[tn]);
      }
    }
    // bias + relu, C layout -> LDS row-major f16 (A layout for layer 2)
    #pragma unroll
    for (int tn = 0; tn < 8; ++tn) {
      int n = tn * 16 + lo;
      float bv = sBe1[n];
      #pragma unroll
      for (int r = 0; r < 8; ++r) {
        float v = c1[tn][r] + bv;
        v = v > 0.f ? v : 0.f;
        h1[(8 * hi + r) * 128 + n] = (_Float16)v;
      }
    }
    asm volatile("s_wait_dscnt 0" ::: "memory");  // cross-lane LDS store->load, same wave
    __builtin_amdgcn_wave_barrier();
    // ---- layer 2: [16x128] @ [128x64] ----
    v8f c2[4];
    #pragma unroll
    for (int tn = 0; tn < 4; ++tn)
      #pragma unroll
      for (int r = 0; r < 8; ++r) c2[tn][r] = 0.f;
    const _Float16* hr = h1 + lo * 128;
    #pragma unroll
    for (int ks = 0; ks < 4; ++ks) {
      int k0 = ks * 32;
      V16 a;
      a.h[0] = *(const v8h*)(hr + k0 + 8 * hi);
      a.h[1] = *(const v8h*)(hr + k0 + 16 + 8 * hi);
      #pragma unroll
      for (int tn = 0; tn < 4; ++tn) {
        const _Float16* wr = We2t + (tn * 16 + lo) * 128 + k0;
        V16 bm;
        bm.h[0] = *(const v8h*)(wr + 8 * hi);
        bm.h[1] = *(const v8h*)(wr + 16 + 8 * hi);
        c2[tn] = wmma_f16(a, bm, c2[tn]);
      }
    }
    // ---- layer 3: bias+relu then 64->1 dot via shfl_xor wave reduction ----
    float part[8];
    #pragma unroll
    for (int r = 0; r < 8; ++r) part[r] = 0.f;
    #pragma unroll
    for (int tn = 0; tn < 4; ++tn) {
      int n = tn * 16 + lo;
      float bv = sBe2[n], w3 = sWe3[n];
      #pragma unroll
      for (int r = 0; r < 8; ++r) {
        float v = c2[tn][r] + bv;
        v = v > 0.f ? v : 0.f;
        part[r] += v * w3;
      }
    }
    #pragma unroll
    for (int r = 0; r < 8; ++r) {
      part[r] += __shfl_xor(part[r], 1, 32);
      part[r] += __shfl_xor(part[r], 2, 32);
      part[r] += __shfl_xor(part[r], 4, 32);
      part[r] += __shfl_xor(part[r], 8, 32);
    }
    if (lo < 8) {                          // 16 writers: m = 8*hi + lo
      float d = part[0];
      #pragma unroll
      for (int r = 1; r < 8; ++r) if (lo == r) d = part[r];
      float w = 1.f / (1.f + __expf(-(d + be3v)));
      int pm = pt * 16 + 8 * hi + lo;
      int i2 = iu[pm], j2 = ju[pm];
      out[(size_t)b * NN * NN + i2 * NN + j2] = w;
      out[(size_t)b * NN * NN + j2 * NN + i2] = w;
    }
    __builtin_amdgcn_wave_barrier();
  }
}

extern "C" void kernel_launch(void* const* d_in, const int* in_sizes, int n_in,
                              void* d_out, int out_size, void* d_ws, size_t ws_size,
                              hipStream_t stream) {
  const float* adj = (const float*)d_in[0];
  const float* emb = (const float*)d_in[1];
  const float* Wg1 = (const float*)d_in[2];  const float* bg1 = (const float*)d_in[3];
  const float* Wg2 = (const float*)d_in[4];  const float* bg2 = (const float*)d_in[5];
  const float* Wg3 = (const float*)d_in[6];  const float* bg3 = (const float*)d_in[7];
  const float* Wmu = (const float*)d_in[8];  const float* bmu = (const float*)d_in[9];
  const float* Wlv = (const float*)d_in[10]; const float* blv = (const float*)d_in[11];
  const float* We1 = (const float*)d_in[12]; const float* be1 = (const float*)d_in[13];
  const float* We2 = (const float*)d_in[14]; const float* be2 = (const float*)d_in[15];
  const float* We3 = (const float*)d_in[16]; const float* be3 = (const float*)d_in[17];
  float* out = (float*)d_out;
  (void)in_sizes; (void)n_in; (void)out_size; (void)ws_size;

  char* ws = (char*)d_ws;
  size_t o = 0;
  _Float16* adjh = (_Float16*)(ws + o); o += (size_t)BB * NN * NN * 2;   // 4 MB
  float*    dinv = (float*)(ws + o);    o += (size_t)BB * NN * 4;        // 16 KB
  _Float16* th   = (_Float16*)(ws + o); o += (size_t)BB * HH * NN * 2;   // 512 KB
  float*    h1   = (float*)(ws + o);    o += (size_t)BB * NN * HH * 4;   // 1 MB
  float*    h2   = (float*)(ws + o);    o += (size_t)BB * NN * HH * 4;
  float*    h3   = (float*)(ws + o);    o += (size_t)BB * NN * HH * 4;
  _Float16* zhb  = (_Float16*)(ws + o); o += (size_t)MU_SZ * 2;          // 256 KB
  _Float16* We1t = (_Float16*)(ws + o); o += (size_t)128 * 64 * 2;
  _Float16* We2t = (_Float16*)(ws + o); o += (size_t)64 * 128 * 2;
  int*      iu   = (int*)(ws + o);      o += (size_t)PAIRS * 4;
  int*      ju   = (int*)(ws + o);      o += (size_t)PAIRS * 4;

  k_pairs<<<NN, NN, 0, stream>>>(iu, ju);
  k_adj_half<<<(ADJ_SZ + 255) / 256, 256, 0, stream>>>(adj, adjh, ADJ_SZ);
  k_deg<<<(BB * NN + 127) / 128, 128, 0, stream>>>(adj, dinv);
  k_wcvt<<<(2 * 128 * 64 + 255) / 256, 256, 0, stream>>>(We1, We2, We1t, We2t);

  int xwN = BB * HH * NN;
  k_xw<<<(xwN + 255) / 256, 256, 0, stream>>>(emb, FF, 0, Wg1, dinv, th);
  k_gcn_agg<<<128, 256, 0, stream>>>(adjh, th, dinv, bg1, nullptr, h1);
  k_xw<<<(xwN + 255) / 256, 256, 0, stream>>>(h1, HH, NN * HH, Wg2, dinv, th);
  k_gcn_agg<<<128, 256, 0, stream>>>(adjh, th, dinv, bg2, h1, h2);
  k_xw<<<(xwN + 255) / 256, 256, 0, stream>>>(h2, HH, NN * HH, Wg3, dinv, th);
  k_gcn_agg<<<128, 256, 0, stream>>>(adjh, th, dinv, bg3, h2, h3);

  k_mu<<<(MU_SZ + 255) / 256, 256, 0, stream>>>(h3, Wmu, bmu, Wlv, blv, out, zhb);
  k_zero_diag<<<(BB * NN + 255) / 256, 256, 0, stream>>>(out);
  k_decode<<<256, 128, 0, stream>>>(zhb, We1t, We2t, be1, be2, We3, be3, iu, ju, out);
}
